// DynamicGraphConv_18184891531673
// MI455X (gfx1250) — matrix operand order repaired and checked by reading, not compiled, not run on hardware
//
#include <hip/hip_runtime.h>
#include <hip/hip_bf16.h>

// ---------------------------------------------------------------------------
// GAT layer on gfx1250: h = xW^T ; e = lrelu(s_i + s_j) ; P = softmax(e) ;
// out = LN(P h). Rank-1 scores => exact row max m_i = lrelu(s_i + max_j s_j),
// single-pass un-normalized accumulation with v_wmma_f32_16x16x32_bf16.
// ---------------------------------------------------------------------------

#define Bb   8
#define Nn   2048
#define Ff   128

typedef __attribute__((ext_vector_type(16))) __bf16 bf16x16;
typedef __attribute__((ext_vector_type(8)))  float  f32x8;

// Pack two f32 -> bf16x2 dword in 3 VALU ops: two round-half-up biases plus
// one v_perm_b32 that merges the high halves (bytes [a2,a3,b2,b3]).
__device__ __forceinline__ unsigned pack2bf(float a, float b) {
    unsigned ua = __builtin_bit_cast(unsigned, a) + 0x8000u;
    unsigned ub = __builtin_bit_cast(unsigned, b) + 0x8000u;
    return __builtin_amdgcn_perm(ub, ua, 0x07060302u);
}

// ---------------------------------------------------------------------------
// Stage 0: fold attention vectors through W:  w_i = W^T a_i, w_j = W^T a_j
// ---------------------------------------------------------------------------
__global__ void prepw_kernel(const float* __restrict__ W,
                             const float* __restrict__ attn,
                             float* __restrict__ wi, float* __restrict__ wj) {
    int f = threadIdx.x;                      // 0..127
    float a = 0.f, c = 0.f;
    #pragma unroll 4
    for (int o = 0; o < Ff; ++o) {
        float wv = W[o * Ff + f];
        a += wv * attn[o];
        c += wv * attn[Ff + o];
    }
    wi[f] = a; wj[f] = c;
}

// ---------------------------------------------------------------------------
// Stage 1: h = x W^T via bf16 WMMA, stored transposed: hT[b][feat][n] (bf16)
// One block = 16 rows of one batch; wave w owns features 16w..16w+15.
// ---------------------------------------------------------------------------
__global__ void proj_h_kernel(const float* __restrict__ x,
                              const float* __restrict__ W,
                              unsigned short* __restrict__ hT) {
    const int blk  = blockIdx.x;              // 0..1023
    const int b    = blk >> 7;
    const int n0   = (blk & 127) << 4;
    const int lane = threadIdx.x & 31;
    const int wv   = threadIdx.x >> 5;        // wave 0..7
    const int col  = lane & 15;
    const bool hi  = lane >= 16;

    const int    arow  = n0 + col;                         // A-matrix row (M)
    const size_t xbase = (size_t)(b * Nn + arow) * Ff;
    const int    akoff = hi ? 16 : 0;
    const int    ocol  = (wv << 4) + col;                  // B-matrix column
    const size_t wbase = (size_t)ocol * Ff + (hi ? 16 : 0);

    f32x8 acc = {};
    #pragma unroll
    for (int f0 = 0; f0 < Ff; f0 += 32) {
        union { bf16x16 v; unsigned d[8]; } A, Bm;
        const float* xp = x + xbase + f0 + akoff;
        #pragma unroll
        for (int i = 0; i < 4; ++i)                       // halves 0..7 : K 0..7 (+base)
            A.d[i] = pack2bf(xp[2 * i], xp[2 * i + 1]);
        #pragma unroll
        for (int i = 0; i < 4; ++i)                       // halves 8..15: K 8..15 (+base)
            A.d[4 + i] = pack2bf(xp[8 + 2 * i], xp[8 + 2 * i + 1]);
        const float* wp = W + wbase + f0;
        #pragma unroll
        for (int i = 0; i < 8; ++i)                       // K consecutive
            Bm.d[i] = pack2bf(wp[2 * i], wp[2 * i + 1]);
        acc = __builtin_amdgcn_wmma_f32_16x16x32_bf16(false, A.v, false, Bm.v,
                                                      (short)0, acc, false, false);
    }

    // D layout: VGPR g -> row M = g (+8 for lanes>=16), col = lane&15.
    // Write transposed: hT[b][feat][n], 8 consecutive n per lane = one b128.
    union { uint4 u; unsigned d[4]; } st;
    #pragma unroll
    for (int g = 0; g < 4; ++g) st.d[g] = pack2bf(acc[2 * g], acc[2 * g + 1]);
    const int feat = (wv << 4) + col;
    const int nb   = n0 + (hi ? 8 : 0);
    *(uint4*)(hT + (size_t)(b * Ff + feat) * Nn + nb) = st.u;
}

// ---------------------------------------------------------------------------
// Stage 1b: s_i[b,n] = x[b,n]·w_i ; s_j likewise (folded projection)
// ---------------------------------------------------------------------------
__global__ void scores_kernel(const float* __restrict__ x,
                              const float* __restrict__ wi,
                              const float* __restrict__ wj,
                              float* __restrict__ si, float* __restrict__ sj) {
    int idx = blockIdx.x * 256 + threadIdx.x;          // 0..16383 (= b*N+n)
    const float* xr = x + (size_t)idx * Ff;
    float a = 0.f, c = 0.f;
    #pragma unroll 8
    for (int f = 0; f < Ff; f += 4) {
        float4 xv = *(const float4*)(xr + f);
        float4 iv = *(const float4*)(wi + f);
        float4 jv = *(const float4*)(wj + f);
        a += xv.x * iv.x + xv.y * iv.y + xv.z * iv.z + xv.w * iv.w;
        c += xv.x * jv.x + xv.y * jv.y + xv.z * jv.z + xv.w * jv.w;
    }
    si[idx] = a; sj[idx] = c;
}

// ---------------------------------------------------------------------------
// Stage 1c: per-batch max of s_j (exact softmax max via monotone lrelu)
// ---------------------------------------------------------------------------
__global__ void maxred_kernel(const float* __restrict__ sj, float* __restrict__ mx) {
    __shared__ float red[256];
    int b = blockIdx.x, t = threadIdx.x;
    float m = -1e30f;
    for (int j = t; j < Nn; j += 256) m = fmaxf(m, sj[b * Nn + j]);
    red[t] = m; __syncthreads();
    for (int s = 128; s > 0; s >>= 1) {
        if (t < s) red[t] = fmaxf(red[t], red[t + s]);
        __syncthreads();
    }
    if (t == 0) mx[b] = red[0];
}

// ---------------------------------------------------------------------------
// Stage 2: out = LN( softmax(lrelu(s_i+s_j)) · h )
// Block = 16 i-rows of one batch; 8 waves split 128 features (16 each).
// j streamed in chunks of 64: P computed once per block in LDS (4 exp/thread),
// each wave does 2 bf16 WMMAs per chunk. B tiles are loaded at the top of the
// iteration so exp/pack/barrier work hides the L2 latency.
// ---------------------------------------------------------------------------
#define PSTRIDE 72   // halves per P row (144B, 16B aligned, bank-spread)

union BfTile { uint4 u[2]; bf16x16 v; };

__global__ void attn_out_kernel(const unsigned short* __restrict__ hT,
                                const float* __restrict__ si_g,
                                const float* __restrict__ sj_g,
                                const float* __restrict__ maxsj,
                                const float* __restrict__ gamma,
                                const float* __restrict__ beta,
                                float* __restrict__ out) {
    __shared__ __align__(16) unsigned short sP[16 * PSTRIDE];
    __shared__ float s_si[16], s_m[16], sZ[256], s_zinv[16];
    __shared__ float s_sum[16][8], s_sq[16][8], s_mu[16], s_rstd[16];

    const int blk  = blockIdx.x;
    const int b    = blk >> 7;
    const int n0   = (blk & 127) << 4;
    const int tid  = threadIdx.x;
    const int lane = tid & 31;
    const int wv   = tid >> 5;
    const int col  = lane & 15;
    const bool hi  = lane >= 16;

    if (tid < 16) {
        float s = si_g[b * Nn + n0 + tid];
        s_si[tid] = s;
        float e = s + maxsj[b];
        s_m[tid] = (e >= 0.f) ? e : 0.2f * e;      // exact row max of lrelu scores
    }
    __syncthreads();

    // P-producer mapping: thread t -> row pr, k = 4*pq .. 4*pq+3
    const int   pr   = tid & 15;
    const int   pq   = tid >> 4;
    const float si_r = s_si[pr];
    const float m_r  = s_m[pr];
    float z_acc = 0.f;

    // WMMA operand geometry
    const int akoff = hi ? 16 : 0;
    const unsigned short* hrow =
        hT + (size_t)(b * Ff + (wv << 4) + col) * Nn + (hi ? 16 : 0);
    const float* sjb = sj_g + b * Nn;

    f32x8 acc = {};
    for (int j0 = 0; j0 < Nn; j0 += 64) {
        // ---- issue this chunk's B-tile loads first (latency hidden below) ----
        BfTile B0, B1;
        {
            const uint4* hp0 = (const uint4*)(hrow + j0);
            const uint4* hp1 = (const uint4*)(hrow + j0 + 32);
            B0.u[0] = hp0[0]; B0.u[1] = hp0[1];
            B1.u[0] = hp1[0]; B1.u[1] = hp1[1];
        }

        // ---- cooperative P tile (16 x 64) -> LDS, plus Z accumulation ----
        {
            float p[4];
            #pragma unroll
            for (int i = 0; i < 4; ++i) {
                float e = si_r + sjb[j0 + 4 * pq + i];
                e = (e >= 0.f) ? e : 0.2f * e;
                p[i] = __expf(e - m_r);
                z_acc += p[i];
            }
            uint2 pk;
            pk.x = pack2bf(p[0], p[1]);
            pk.y = pack2bf(p[2], p[3]);
            *(uint2*)&sP[pr * PSTRIDE + 4 * pq] = pk;
        }
        __syncthreads();

        // ---- two 16x16x32 bf16 WMMAs over this 64-wide chunk ----
        BfTile A0, A1;
        A0.u[0] = *(const uint4*)&sP[col * PSTRIDE + akoff];
        A0.u[1] = *(const uint4*)&sP[col * PSTRIDE + 8 + akoff];
        A1.u[0] = *(const uint4*)&sP[col * PSTRIDE + 32 + akoff];
        A1.u[1] = *(const uint4*)&sP[col * PSTRIDE + 40 + akoff];
        acc = __builtin_amdgcn_wmma_f32_16x16x32_bf16(false, A0.v, false, B0.v,
                                                      (short)0, acc, false, false);
        acc = __builtin_amdgcn_wmma_f32_16x16x32_bf16(false, A1.v, false, B1.v,
                                                      (short)0, acc, false, false);
        __syncthreads();   // protect sP before next producer pass
    }

    // ---- Z reduction: Z[r] = sum over the 16 producer columns ----
    sZ[tid] = z_acc;
    __syncthreads();
    if (tid < 16) {
        float z = 0.f;
        #pragma unroll
        for (int q = 0; q < 16; ++q) z += sZ[q * 16 + tid];
        s_zinv[tid] = 1.0f / z;
    }
    __syncthreads();

    // ---- normalize by Z, then LayerNorm across the 128 features ----
    const int rbase = hi ? 8 : 0;
    float dn[8];
    #pragma unroll
    for (int g = 0; g < 8; ++g) dn[g] = acc[g] * s_zinv[g + rbase];

    #pragma unroll
    for (int g = 0; g < 8; ++g) {
        float t1 = dn[g], t2 = dn[g] * dn[g];
        #pragma unroll
        for (int m = 1; m < 16; m <<= 1) {          // reduce 16 cols per half-wave
            t1 += __shfl_xor(t1, m, 16);
            t2 += __shfl_xor(t2, m, 16);
        }
        if (col == 0) { s_sum[g + rbase][wv] = t1; s_sq[g + rbase][wv] = t2; }
    }
    __syncthreads();
    if (tid < 16) {
        float su = 0.f, sq = 0.f;
        #pragma unroll
        for (int q = 0; q < 8; ++q) { su += s_sum[tid][q]; sq += s_sq[tid][q]; }
        float mu  = su * (1.0f / 128.0f);
        float var = sq * (1.0f / 128.0f) - mu * mu;
        s_mu[tid]   = mu;
        s_rstd[tid] = rsqrtf(var + 1e-5f);
    }
    __syncthreads();

    const int   feat = (wv << 4) + col;
    const float gm = gamma[feat], bt = beta[feat];
    float* orow = out + (size_t)(b * Nn + n0 + rbase) * Ff + feat;
    #pragma unroll
    for (int g = 0; g < 8; ++g) {
        int rg = g + rbase;
        orow[(size_t)g * Ff] = (dn[g] - s_mu[rg]) * s_rstd[rg] * gm + bt;
    }
}

// ---------------------------------------------------------------------------
// Host launcher
// ---------------------------------------------------------------------------
extern "C" void kernel_launch(void* const* d_in, const int* in_sizes, int n_in,
                              void* d_out, int out_size, void* d_ws, size_t ws_size,
                              hipStream_t stream) {
    const float* x     = (const float*)d_in[0];
    const float* W     = (const float*)d_in[1];
    const float* attn  = (const float*)d_in[2];
    const float* gamma = (const float*)d_in[3];
    const float* beta  = (const float*)d_in[4];
    float*       outp  = (float*)d_out;

    char* ws = (char*)d_ws;
    unsigned short* hT = (unsigned short*)ws;                    // 4 MiB bf16 h^T
    float* s_i  = (float*)(ws + 4194304);                        // 64 KiB
    float* s_j  = (float*)(ws + 4194304 + 65536);                // 64 KiB
    float* w_i  = (float*)(ws + 4194304 + 2 * 65536);            // 512 B
    float* w_j  = w_i + Ff;
    float* mxsj = w_j + Ff;

    prepw_kernel   <<<1,               128, 0, stream>>>(W, attn, w_i, w_j);
    proj_h_kernel  <<<Bb * (Nn / 16),  256, 0, stream>>>(x, W, hT);
    scores_kernel  <<<Bb * Nn / 256,   256, 0, stream>>>(x, w_i, w_j, s_i, s_j);
    maxred_kernel  <<<Bb,              256, 0, stream>>>(s_j, mxsj);
    attn_out_kernel<<<Bb * (Nn / 16),  256, 0, stream>>>(hT, s_i, s_j, mxsj,
                                                         gamma, beta, outp);
}
